// LSTMCellModel_27960237097158
// MI455X (gfx1250) — matrix operand order repaired and checked by reading, not compiled, not run on hardware
//
#include <hip/hip_runtime.h>
#include <hip/hip_bf16.h>

#define Bdim   256
#define Tdim   512
#define INdim  9
#define Hdim   512
#define OUTdim 3

#define NWG     32
#define HS      16
#define THREADS 256
#define KX      32          // extra K-tile width holding [xin(9), 1.0, pad]

typedef __bf16 v16bf __attribute__((ext_vector_type(16)));
typedef __bf16 v8bf  __attribute__((ext_vector_type(8)));
typedef float  v8f   __attribute__((ext_vector_type(8)));

__device__ __forceinline__ unsigned short f2bf(float f) {
  unsigned u = __builtin_bit_cast(unsigned, f);
  u += 0x7FFFu + ((u >> 16) & 1u);
  return (unsigned short)(u >> 16);
}
__device__ __forceinline__ float sigm(float x)   { return 1.0f / (1.0f + __expf(-x)); }
__device__ __forceinline__ float tanhfst(float x){ return 2.0f / (1.0f + __expf(-2.0f * x)) - 1.0f; }

// ---------------------------------------------------------------------------
// Init: out[b,t,:] = b_fc (fc partials accumulate on top), zero hx ping-pong
// buffers and barrier counter. Runs every launch -> deterministic under replay.
// ---------------------------------------------------------------------------
__global__ void lstm_init_kernel(float* __restrict__ out, const float* __restrict__ b_fc,
                                 unsigned short* __restrict__ hxg, unsigned* __restrict__ ctr) {
  int i = blockIdx.x * blockDim.x + threadIdx.x;
  int stride = gridDim.x * blockDim.x;
  if (i == 0) *ctr = 0u;
  float b0 = b_fc[0], b1 = b_fc[1], b2 = b_fc[2];
  for (int idx = i; idx < Bdim * Tdim; idx += stride) {
    float* p = out + (size_t)idx * OUTdim;
    p[0] = b0; p[1] = b1; p[2] = b2;
  }
  for (int idx = i; idx < 2 * Bdim * Hdim; idx += stride) hxg[idx] = 0;
}

// ---------------------------------------------------------------------------
// Persistent LSTM: WG w owns hidden units [16w,16w+16); wave v owns batch rows
// [32v,32v+32). All GEMM work (recurrent, input, biases) on the WMMA pipe.
// ---------------------------------------------------------------------------
__global__ __launch_bounds__(THREADS)
void lstm_persistent(const float* __restrict__ x,      // [B,T,IN]
                     const float* __restrict__ W_ih,   // [4H,IN]
                     const float* __restrict__ W_hh,   // [4H,H]
                     const float* __restrict__ b_ih,   // [4H]
                     const float* __restrict__ b_hh,   // [4H]
                     const float* __restrict__ W_fc,   // [OUT,H]
                     const int*   __restrict__ horizon_p,
                     float* __restrict__ out,          // [B,T,OUT] pre-filled with b_fc
                     unsigned* __restrict__ ctr,
                     unsigned short* __restrict__ hxg) // bf16 [2][B][H]
{
  __shared__ unsigned short ldsW [4 * HS * Hdim];  // 64 KB: W_hh slice [g][n][k]
  __shared__ unsigned short ldsWx[4 * HS * KX];    //  4 KB: [W_ih(9) | b_ih+b_hh | 0...]
  __shared__ unsigned short ldsX [Bdim * KX];      // 16 KB: per-step [xin(9) | 1.0 | 0...]

  const int wg   = blockIdx.x;
  const int tid  = threadIdx.x;
  const int wave = tid >> 5;
  const int lane = tid & 31;
  const int hlf  = lane >> 4;
  const int nloc = lane & 15;
  const int j    = wg * HS + nloc;      // this lane's hidden unit in C/D layout
  const int hor  = horizon_p[0];

  // --- one-time LDS staging -------------------------------------------------
  for (int idx = tid; idx < 4 * HS * Hdim; idx += THREADS) {
    int rl = idx >> 9;                  // g*16+n   (Hdim==512)
    int k  = idx & (Hdim - 1);
    int g  = rl >> 4, n = rl & 15;
    int grow = g * Hdim + wg * HS + n;
    ldsW[idx] = f2bf(W_hh[(size_t)grow * Hdim + k]);
  }
  for (int idx = tid; idx < 4 * HS * KX; idx += THREADS) {
    int rl = idx >> 5;                  // g*16+n   (KX==32)
    int kk = idx & (KX - 1);
    int g = rl >> 4, n = rl & 15;
    int grow = g * Hdim + wg * HS + n;
    float v = 0.0f;
    if (kk < INdim)       v = W_ih[(size_t)grow * INdim + kk];
    else if (kk == INdim) v = b_ih[grow] + b_hh[grow];
    ldsWx[idx] = f2bf(v);
  }
  for (int idx = tid; idx < Bdim * KX; idx += THREADS) ldsX[idx] = 0;
  for (int m = tid; m < Bdim; m += THREADS)
    ldsX[m * KX + INdim] = (unsigned short)0x3F80;   // bf16 1.0 -> bias column

  float wfc[OUTdim];
  #pragma unroll
  for (int o = 0; o < OUTdim; ++o) wfc[o] = W_fc[o * Hdim + j];

  __syncthreads();

  // cell state in registers: (mt,r) <-> m = wave*32 + mt*16 + hlf*8 + r
  float cst[2][8];
  #pragma unroll
  for (int mt = 0; mt < 2; ++mt)
    #pragma unroll
    for (int r = 0; r < 8; ++r) cst[mt][r] = 0.0f;

  const int m0w = wave * 32;

  for (int t = 0; t < Tdim; ++t) {
    const unsigned short* hin  = hxg + (size_t)(t & 1)       * (Bdim * Hdim);
    unsigned short*       hout = hxg + (size_t)((t + 1) & 1) * (Bdim * Hdim);
    const bool raw = (t < 5) || (t % hor == 0);

    // ---- stage xin row for this step: thread m handles batch row m --------
    {
      const int m = tid;                // THREADS == Bdim
      const float* xp = x + ((size_t)m * Tdim + t) * INdim;
      float v0 = xp[0], v1 = xp[1], v2 = xp[2];
      if (!raw) {
        const float* po = out + ((size_t)m * Tdim + (t - 1)) * OUTdim;
        v0 = po[0]; v1 = po[1]; v2 = po[2];
      }
      unsigned short* xr = &ldsX[m * KX];
      xr[0] = f2bf(v0); xr[1] = f2bf(v1); xr[2] = f2bf(v2);
      #pragma unroll
      for (int q = 3; q < INdim; ++q) xr[q] = f2bf(xp[q]);
    }
    __syncthreads();                    // ldsX visible to all waves

    // ---- GEMM: zero-init accumulators, everything via WMMA ----------------
    v8f acc[2][4];
    #pragma unroll
    for (int mt = 0; mt < 2; ++mt)
      #pragma unroll
      for (int g = 0; g < 4; ++g) acc[mt][g] = (v8f){0.f,0.f,0.f,0.f,0.f,0.f,0.f,0.f};

    const int arow0 = m0w + nloc;       // A layout: M = lane&15
    const int arow1 = m0w + 16 + nloc;

    // recurrent part: hx(t-1)[256,512] @ W_hh_slice^T
    for (int k0 = 0; k0 < Hdim; k0 += 32) {
      v16bf Afrag0, Afrag1;
      {
        const unsigned short* pa0 = hin + (size_t)arow0 * Hdim + k0 + hlf * 8;
        const unsigned short* pa1 = hin + (size_t)arow1 * Hdim + k0 + hlf * 8;
        v8bf lo0 = *(const v8bf*)pa0;
        v8bf hi0 = *(const v8bf*)(pa0 + 16);
        v8bf lo1 = *(const v8bf*)pa1;
        v8bf hi1 = *(const v8bf*)(pa1 + 16);
        #pragma unroll
        for (int e = 0; e < 8; ++e) {
          Afrag0[e] = lo0[e]; Afrag0[e + 8] = hi0[e];
          Afrag1[e] = lo1[e]; Afrag1[e + 8] = hi1[e];
        }
      }
      #pragma unroll
      for (int g = 0; g < 4; ++g) {
        const unsigned short* pb = &ldsW[(g * HS + nloc) * Hdim + k0 + hlf * 8];
        v16bf Bfrag = __builtin_shufflevector(*(const v8bf*)pb, *(const v8bf*)(pb + 16), 0,1,2,3,4,5,6,7,8,9,10,11,12,13,14,15);
        acc[0][g] = __builtin_amdgcn_wmma_f32_16x16x32_bf16(
            false, Afrag0, false, Bfrag, (short)0, acc[0][g], false, false);
        acc[1][g] = __builtin_amdgcn_wmma_f32_16x16x32_bf16(
            false, Afrag1, false, Bfrag, (short)0, acc[1][g], false, false);
      }
    }

    // input + bias part: one extra K-tile from LDS ([xin|1|pad] @ [W_ih|b|0]^T)
    {
      v16bf Ax0, Ax1;
      {
        const unsigned short* pa0 = &ldsX[arow0 * KX + hlf * 8];
        const unsigned short* pa1 = &ldsX[arow1 * KX + hlf * 8];
        v8bf lo0 = *(const v8bf*)pa0;
        v8bf hi0 = *(const v8bf*)(pa0 + 16);
        v8bf lo1 = *(const v8bf*)pa1;
        v8bf hi1 = *(const v8bf*)(pa1 + 16);
        #pragma unroll
        for (int e = 0; e < 8; ++e) {
          Ax0[e] = lo0[e]; Ax0[e + 8] = hi0[e];
          Ax1[e] = lo1[e]; Ax1[e + 8] = hi1[e];
        }
      }
      #pragma unroll
      for (int g = 0; g < 4; ++g) {
        const unsigned short* pb = &ldsWx[(g * HS + nloc) * KX + hlf * 8];
        v16bf Bfrag = __builtin_shufflevector(*(const v8bf*)pb, *(const v8bf*)(pb + 16), 0,1,2,3,4,5,6,7,8,9,10,11,12,13,14,15);
        acc[0][g] = __builtin_amdgcn_wmma_f32_16x16x32_bf16(
            false, Ax0, false, Bfrag, (short)0, acc[0][g], false, false);
        acc[1][g] = __builtin_amdgcn_wmma_f32_16x16x32_bf16(
            false, Ax1, false, Bfrag, (short)0, acc[1][g], false, false);
      }
    }

    // ---- gates -> state -> publish hx(t) bf16 -> fc partials --------------
    for (int mt = 0; mt < 2; ++mt) {
      #pragma unroll
      for (int r = 0; r < 8; ++r) {
        int m = m0w + mt * 16 + hlf * 8 + r;
        float iv = acc[mt][0][r], fv = acc[mt][1][r];
        float gv = acc[mt][2][r], ov = acc[mt][3][r];
        float c2 = sigm(fv) * cst[mt][r] + sigm(iv) * tanhfst(gv);
        float h2 = sigm(ov) * tanhfst(c2);
        cst[mt][r] = c2;
        hout[(size_t)m * Hdim + j] = f2bf(h2);
        #pragma unroll
        for (int o = 0; o < OUTdim; ++o) {
          float p = h2 * wfc[o];
          p += __shfl_xor(p, 1, 16);
          p += __shfl_xor(p, 2, 16);
          p += __shfl_xor(p, 4, 16);
          p += __shfl_xor(p, 8, 16);
          if (nloc == 0)
            atomicAdd(&out[((size_t)m * Tdim + t) * OUTdim + o], p);
        }
      }
    }

    // ---- grid-wide barrier (monotonic counter, agent-scope fences) --------
    __builtin_amdgcn_fence(__ATOMIC_RELEASE, "agent");
    __syncthreads();
    if (tid == 0) {
      atomicAdd(ctr, 1u);
      const unsigned target = (unsigned)(t + 1) * NWG;
      while (__hip_atomic_load(ctr, __ATOMIC_RELAXED, __HIP_MEMORY_SCOPE_AGENT) < target)
        __builtin_amdgcn_s_sleep(1);
    }
    __syncthreads();
    __builtin_amdgcn_fence(__ATOMIC_ACQUIRE, "agent");
  }
}

extern "C" void kernel_launch(void* const* d_in, const int* in_sizes, int n_in,
                              void* d_out, int out_size, void* d_ws, size_t ws_size,
                              hipStream_t stream) {
  const float* x    = (const float*)d_in[0];
  const float* W_ih = (const float*)d_in[1];
  const float* W_hh = (const float*)d_in[2];
  const float* b_ih = (const float*)d_in[3];
  const float* b_hh = (const float*)d_in[4];
  const float* W_fc = (const float*)d_in[5];
  const float* b_fc = (const float*)d_in[6];
  const int* horizon = (const int*)d_in[7];
  float* out = (float*)d_out;

  unsigned* ctr = (unsigned*)d_ws;                               // 256 B reserved
  unsigned short* hxg = (unsigned short*)((char*)d_ws + 256);    // 2*256*512 bf16 = 512 KB

  lstm_init_kernel<<<512, 256, 0, stream>>>(out, b_fc, hxg, ctr);
  lstm_persistent<<<NWG, THREADS, 0, stream>>>(x, W_ih, W_hh, b_ih, b_hh, W_fc,
                                               horizon, out, ctr, hxg);
}
